// Emu3Attention_26611617366710
// MI455X (gfx1250) — compile-verified, matrix operand order checked by reading, not run on hardware
//
#include <hip/hip_runtime.h>
#include <hip/hip_bf16.h>
#include <math.h>

typedef __attribute__((ext_vector_type(16))) __bf16 v16bf;
typedef __attribute__((ext_vector_type(8)))  __bf16 v8bf;
typedef __attribute__((ext_vector_type(8)))  float  v8f;

#define SEQ      2048
#define HIDDEN   4096
#define NHEADS   32
#define NKV      8
#define HD       128
#define KDIM     1024   // NKV*HD

// ---------------------------------------------------------------------------
// bf16 fragment loader. CDNA5 16-bit A/B fragment layout (ISA 7.12.2):
//   lanes 0-15 : row = lane,    K = {0..7}  then {16..23}
//   lanes 16-31: row = lane-16, K = {8..15} then {24..31}
// Source is row-major [row][K] with leading dim `ld`; 16B-aligned tiles.
// ---------------------------------------------------------------------------
__device__ __forceinline__ v16bf load_frag_bf16(const __bf16* __restrict__ base,
                                                int ld, int lane) {
  const int half = lane >> 4;
  const int r    = lane & 15;
  const __bf16* p = base + (size_t)r * ld + half * 8;
  v8bf lo = *(const v8bf*)p;
  v8bf hi = *(const v8bf*)(p + 16);
  return __builtin_shufflevector(lo, hi, 0,1,2,3,4,5,6,7,8,9,10,11,12,13,14,15);
}

// ---------------------------------------------------------------------------
// One-time fp32 -> bf16 staging (8 elements/thread, n % 8 == 0).
// ---------------------------------------------------------------------------
__global__ void cvt_f32_bf16(const float* __restrict__ src,
                             __bf16* __restrict__ dst, int n) {
  int i = (blockIdx.x * blockDim.x + threadIdx.x) * 8;
  if (i >= n) return;
  float4 a = *(const float4*)(src + i);
  float4 b = *(const float4*)(src + i + 4);
  v8bf o;
  o[0] = (__bf16)a.x; o[1] = (__bf16)a.y; o[2] = (__bf16)a.z; o[3] = (__bf16)a.w;
  o[4] = (__bf16)b.x; o[5] = (__bf16)b.y; o[6] = (__bf16)b.z; o[7] = (__bf16)b.w;
  *(v8bf*)(dst + i) = o;
}

// ---------------------------------------------------------------------------
// GEMM: C[M,N] = A[M,K] * B[N,K]^T   (bf16 operands, f32 WMMA accumulate)
// block = 256 threads (8 waves) -> block tile 128x256, wave tile 64x64
// (4x4 grid of 16x16 WMMA accumulators = 128 acc VGPRs; 16 WMMA per K-step
// against 16 global_load_b128). grid = (N/256, M/128).
// ---------------------------------------------------------------------------
template <bool OUT_F32>
__global__ __launch_bounds__(256) void gemm_bf16_wmma(
    const __bf16* __restrict__ A, const __bf16* __restrict__ B,
    void* __restrict__ Cout, int M, int N, int K) {
  const int wave = threadIdx.x >> 5;
  const int lane = threadIdx.x & 31;
  const int wm   = wave >> 2;             // 0..1 (64 rows each)
  const int wn   = wave & 3;              // 0..3 (64 cols each)
  const int m0   = blockIdx.y * 128 + wm * 64;
  const int n0   = blockIdx.x * 256 + wn * 64;
  const int half = lane >> 4;
  const int l16  = lane & 15;

  v8f acc[4][4];
#pragma unroll
  for (int i = 0; i < 4; ++i)
#pragma unroll
    for (int j = 0; j < 4; ++j) acc[i][j] = (v8f){0.f,0.f,0.f,0.f,0.f,0.f,0.f,0.f};

  for (int k0 = 0; k0 < K; k0 += 32) {
    if (k0 + 32 < K) {  // global_prefetch_b8 on the next K tile
      __builtin_prefetch(A + (size_t)m0 * K + k0 + 32, 0, 1);
      __builtin_prefetch(B + (size_t)n0 * K + k0 + 32, 0, 1);
    }
    v16bf a[4], b[4];
#pragma unroll
    for (int i = 0; i < 4; ++i)
      a[i] = load_frag_bf16(A + (size_t)(m0 + i * 16) * K + k0, K, lane);
#pragma unroll
    for (int j = 0; j < 4; ++j)
      b[j] = load_frag_bf16(B + (size_t)(n0 + j * 16) * K + k0, K, lane);
#pragma unroll
    for (int i = 0; i < 4; ++i)
#pragma unroll
      for (int j = 0; j < 4; ++j)
        acc[i][j] = __builtin_amdgcn_wmma_f32_16x16x32_bf16(
            false, a[i], false, b[j], (short)0, acc[i][j], false, false);
  }

  // C layout: VGPR r -> row r (lanes 0-15) / r+8 (lanes 16-31), col = l16
#pragma unroll
  for (int i = 0; i < 4; ++i)
#pragma unroll
    for (int j = 0; j < 4; ++j)
#pragma unroll
      for (int r = 0; r < 8; ++r) {
        const size_t row = m0 + i * 16 + r + 8 * half;
        const size_t col = n0 + j * 16 + l16;
        if (OUT_F32) ((float*)Cout)[row * N + col] = acc[i][j][r];
        else         ((__bf16*)Cout)[row * N + col] = (__bf16)acc[i][j][r];
      }
}

// ---------------------------------------------------------------------------
// RoPE in place on bf16 (fp32 math). X: [SEQ, nheads*HD]; pos = row index.
// ---------------------------------------------------------------------------
__global__ void rope_bf16(__bf16* __restrict__ X, int nheads) {
  int tid = blockIdx.x * blockDim.x + threadIdx.x;
  int total = SEQ * nheads * (HD / 2);
  if (tid >= total) return;
  int d   = tid % (HD / 2);
  int h   = (tid / (HD / 2)) % nheads;
  int pos = tid / ((HD / 2) * nheads);
  float inv_freq = __powf(1000000.0f, -(float)(2 * d) / (float)HD);
  float ang = (float)pos * inv_freq;
  float c = __cosf(ang), s = __sinf(ang);
  size_t base = (size_t)pos * nheads * HD + (size_t)h * HD;
  float x1 = (float)X[base + d];
  float x2 = (float)X[base + d + HD / 2];
  X[base + d]          = (__bf16)(x1 * c - x2 * s);
  X[base + d + HD / 2] = (__bf16)(x2 * c + x1 * s);
}

// V [SEQ, KDIM] bf16 -> Vt [NKV][HD][SEQ] bf16 (per-head transposed)
__global__ void v_transpose(const __bf16* __restrict__ V, __bf16* __restrict__ Vt) {
  int tid = blockIdx.x * blockDim.x + threadIdx.x;
  if (tid >= SEQ * KDIM) return;
  int col = tid % KDIM;
  int n   = tid / KDIM;
  int h   = col >> 7;
  int d   = col & 127;
  Vt[(size_t)h * HD * SEQ + (size_t)d * SEQ + n] = V[(size_t)n * KDIM + col];
}

// ---------------------------------------------------------------------------
// Flash attention (causal, GQA 4:1). One wave = 16 query rows, one block =
// 8 waves = 128 query rows of one head. grid = (SEQ/128, NHEADS).
// Qb:[SEQ,HIDDEN], Kb:[SEQ,KDIM] (both rope'd bf16), Vt:[NKV][HD][SEQ] bf16,
// Ab:[SEQ,HIDDEN] bf16 output.
// ---------------------------------------------------------------------------
__global__ __launch_bounds__(256) void attn_wmma(
    const __bf16* __restrict__ Qb, const __bf16* __restrict__ Kb,
    const __bf16* __restrict__ Vt, __bf16* __restrict__ Ab) {
  const int h    = blockIdx.y;
  const int hkv  = h >> 2;
  const int wave = threadIdx.x >> 5;
  const int lane = threadIdx.x & 31;
  const int half = lane >> 4;
  const int l16  = lane & 15;
  const int m0   = blockIdx.x * 128 + wave * 16;

  __shared__ __align__(16) __bf16 lds_p[8][16 * 32];
  __bf16* myP = &lds_p[wave][0];

  v8f acc[8];
#pragma unroll
  for (int t = 0; t < 8; ++t) acc[t] = (v8f){0.f,0.f,0.f,0.f,0.f,0.f,0.f,0.f};
  float mrow[8], lrow[8];
#pragma unroll
  for (int r = 0; r < 8; ++r) { mrow[r] = -1e30f; lrow[r] = 0.f; }

  const float scale = 0.08838834764831845f;  // 1/sqrt(128)
  const __bf16* Vh = Vt + (size_t)hkv * HD * SEQ;
  const int kv_end = m0 + 16;                // causal bound for this tile

  for (int kb = 0; kb < kv_end; kb += 32) {
    // ---- S = Q K^T (two 16x16 tiles over keys kb..kb+31; contraction d=128)
    v8f s0 = (v8f){0.f,0.f,0.f,0.f,0.f,0.f,0.f,0.f};
    v8f s1 = (v8f){0.f,0.f,0.f,0.f,0.f,0.f,0.f,0.f};
#pragma unroll
    for (int kk = 0; kk < 4; ++kk) {
      v16bf a  = load_frag_bf16(Qb + (size_t)m0 * HIDDEN + h * HD + kk * 32, HIDDEN, lane);
      v16bf b0 = load_frag_bf16(Kb + (size_t)kb * KDIM + hkv * HD + kk * 32, KDIM, lane);
      v16bf b1 = load_frag_bf16(Kb + (size_t)(kb + 16) * KDIM + hkv * HD + kk * 32, KDIM, lane);
      s0 = __builtin_amdgcn_wmma_f32_16x16x32_bf16(false, a, false, b0, (short)0, s0, false, false);
      s1 = __builtin_amdgcn_wmma_f32_16x16x32_bf16(false, a, false, b1, (short)0, s1, false, false);
    }

    // ---- online softmax (row stats per 16-lane column group)
#pragma unroll
    for (int r = 0; r < 8; ++r) {
      const int q  = m0 + r + 8 * half;
      float v0 = s0[r] * scale;
      float v1 = s1[r] * scale;
      if (kb + l16 > q)      v0 = -1e30f;
      if (kb + 16 + l16 > q) v1 = -1e30f;

      float mx = fmaxf(v0, v1);
#pragma unroll
      for (int off = 8; off >= 1; off >>= 1) mx = fmaxf(mx, __shfl_xor(mx, off, 32));
      const float mnew = fmaxf(mrow[r], mx);
      const float corr = __expf(mrow[r] - mnew);
      const float p0 = __expf(v0 - mnew);
      const float p1 = __expf(v1 - mnew);
      float rs = p0 + p1;
#pragma unroll
      for (int off = 8; off >= 1; off >>= 1) rs += __shfl_xor(rs, off, 32);
      lrow[r] = lrow[r] * corr + rs;
      mrow[r] = mnew;
#pragma unroll
      for (int t = 0; t < 8; ++t) acc[t][r] *= corr;

      // stage P to LDS in C-tile order for re-layout to A-fragment order
      myP[(r + 8 * half) * 32 + l16]      = (__bf16)p0;
      myP[(r + 8 * half) * 32 + 16 + l16] = (__bf16)p1;
    }
    asm volatile("s_wait_dscnt 0" ::: "memory");  // wave-local LDS store->load

    // ---- reload P as an A fragment (16x32)
    const __bf16* pp = myP + l16 * 32 + half * 8;
    v8bf lo = *(const v8bf*)pp;
    v8bf hi = *(const v8bf*)(pp + 16);
    v16bf pa = __builtin_shufflevector(lo, hi, 0,1,2,3,4,5,6,7,8,9,10,11,12,13,14,15);

    // ---- O += P @ V  (B fragment from Vt: row=d, K=key index, contiguous)
#pragma unroll
    for (int dt = 0; dt < 8; ++dt) {
      v16bf bv = load_frag_bf16(Vh + (size_t)(dt * 16) * SEQ + kb, SEQ, lane);
      acc[dt] = __builtin_amdgcn_wmma_f32_16x16x32_bf16(
          false, pa, false, bv, (short)0, acc[dt], false, false);
    }
  }

  // ---- epilogue: normalize, write bf16 [SEQ, HIDDEN]
#pragma unroll
  for (int dt = 0; dt < 8; ++dt)
#pragma unroll
    for (int r = 0; r < 8; ++r) {
      const int q = m0 + r + 8 * half;
      const int d = dt * 16 + l16;
      Ab[(size_t)q * HIDDEN + h * HD + d] = (__bf16)(acc[dt][r] / lrow[r]);
    }
}

// ---------------------------------------------------------------------------
extern "C" void kernel_launch(void* const* d_in, const int* in_sizes, int n_in,
                              void* d_out, int out_size, void* d_ws, size_t ws_size,
                              hipStream_t stream) {
  const float* hidden = (const float*)d_in[0];
  // d_in[1] = position_ids (arange; positions derived from row index)
  const float* Wq = (const float*)d_in[2];
  const float* Wk = (const float*)d_in[3];
  const float* Wv = (const float*)d_in[4];
  const float* Wo = (const float*)d_in[5];
  float* out = (float*)d_out;

  char* ws = (char*)d_ws;
  const size_t MB = (size_t)1 << 20;
  __bf16* Xb  = (__bf16*)(ws);             // 16 MB  [SEQ, HIDDEN]
  __bf16* Wqb = (__bf16*)(ws +  16 * MB);  // 32 MB  [HIDDEN, HIDDEN]
  __bf16* Wkb = (__bf16*)(ws +  48 * MB);  //  8 MB  [KDIM, HIDDEN]
  __bf16* Wvb = (__bf16*)(ws +  56 * MB);  //  8 MB  [KDIM, HIDDEN]
  __bf16* Wob = (__bf16*)(ws +  64 * MB);  // 32 MB  [HIDDEN, HIDDEN]
  __bf16* Qb  = (__bf16*)(ws +  96 * MB);  // 16 MB  [SEQ, HIDDEN]
  __bf16* Kb  = (__bf16*)(ws + 112 * MB);  //  4 MB  [SEQ, KDIM]
  __bf16* Vb  = (__bf16*)(ws + 116 * MB);  //  4 MB  [SEQ, KDIM]
  __bf16* Vt  = (__bf16*)(ws + 120 * MB);  //  4 MB  [NKV, HD, SEQ]
  __bf16* Ab  = (__bf16*)(ws + 124 * MB);  // 16 MB  [SEQ, HIDDEN] (tot 140 MB)

  dim3 blk(256);

  // One-time bf16 staging of activations and weights
  cvt_f32_bf16<<<(SEQ * HIDDEN / 8 + 255) / 256, blk, 0, stream>>>(hidden, Xb, SEQ * HIDDEN);
  cvt_f32_bf16<<<(HIDDEN * HIDDEN / 8 + 255) / 256, blk, 0, stream>>>(Wq, Wqb, HIDDEN * HIDDEN);
  cvt_f32_bf16<<<(KDIM * HIDDEN / 8 + 255) / 256, blk, 0, stream>>>(Wk, Wkb, KDIM * HIDDEN);
  cvt_f32_bf16<<<(KDIM * HIDDEN / 8 + 255) / 256, blk, 0, stream>>>(Wv, Wvb, KDIM * HIDDEN);
  cvt_f32_bf16<<<(HIDDEN * HIDDEN / 8 + 255) / 256, blk, 0, stream>>>(Wo, Wob, HIDDEN * HIDDEN);

  // Projections: X @ W^T (bf16 in, bf16 out)
  gemm_bf16_wmma<false><<<dim3(HIDDEN / 256, SEQ / 128), blk, 0, stream>>>(Xb, Wqb, Qb, SEQ, HIDDEN, HIDDEN);
  gemm_bf16_wmma<false><<<dim3(KDIM   / 256, SEQ / 128), blk, 0, stream>>>(Xb, Wkb, Kb, SEQ, KDIM,   HIDDEN);
  gemm_bf16_wmma<false><<<dim3(KDIM   / 256, SEQ / 128), blk, 0, stream>>>(Xb, Wvb, Vb, SEQ, KDIM,   HIDDEN);

  // RoPE (in place) + V transpose
  {
    int nq = SEQ * NHEADS * (HD / 2);
    rope_bf16<<<(nq + 255) / 256, blk, 0, stream>>>(Qb, NHEADS);
    int nk = SEQ * NKV * (HD / 2);
    rope_bf16<<<(nk + 255) / 256, blk, 0, stream>>>(Kb, NKV);
    int nv = SEQ * KDIM;
    v_transpose<<<(nv + 255) / 256, blk, 0, stream>>>(Vb, Vt);
  }

  // Causal GQA flash attention -> bf16
  attn_wmma<<<dim3(SEQ / 128, NHEADS), blk, 0, stream>>>(Qb, Kb, Vt, Ab);

  // Output projection: Ab @ Wo^T -> fp32 d_out
  gemm_bf16_wmma<true><<<dim3(HIDDEN / 256, SEQ / 128), blk, 0, stream>>>(Ab, Wob, out, SEQ, HIDDEN, HIDDEN);
}
